// DoubleXLSTMDown_58557584113768
// MI455X (gfx1250) — matrix operand order repaired and checked by reading, not compiled, not run on hardware
//
#include <hip/hip_runtime.h>
#include <hip/hip_bf16.h>
#include <math.h>

// ---------------- problem constants ----------------
#define B_    8
#define S_    1024
#define E_    512
#define L_    2
#define H_    4
#define KCONV 4
#define F_    128
#define I_    1024          // 2*E
#define TWOI  2048
#define DH_   256
#define BS_   (B_ * S_)     // 8192

typedef __bf16 bf16;
typedef __attribute__((ext_vector_type(16))) __bf16    v16bf;
typedef __attribute__((ext_vector_type(8)))  float     v8f;
typedef __attribute__((ext_vector_type(4)))  unsigned  uv4;

// ---------------- WMMA helper (CDNA5 16x16x32 bf16, fp32 acc) ----------------
__device__ inline v8f wmma_bf(v16bf a, v16bf b, v8f c) {
  return __builtin_amdgcn_wmma_f32_16x16x32_bf16(false, a, false, b, (short)0, c, false, false);
}

// A fragment 16x32 bf16 (ISA 7.12.2): per lane two contiguous 16B runs -> 2x ds_load_b128.
// Requires ldm % 8 == 0 and k0 % 8 == 0.
__device__ inline v16bf load_a_frag(const bf16* __restrict__ base, int row0, int k0,
                                    int ldm, int lane) {
  int m = lane & 15, lh = lane >> 4;
  const bf16* p = base + (size_t)(row0 + m) * ldm + k0 + lh * 16;
  union { uv4 q[2]; v16bf v; } u;
  u.q[0] = *(const uv4*)p;         // K = lh*16 .. lh*16+7
  u.q[1] = *(const uv4*)(p + 8);   // K = lh*16+8 .. lh*16+15
  return u.v;
}

// B fragment 32x16 from K-major transposed-pair u32 LDS:
//   word[n * ldU + kp] = {B[2kp][n], B[2kp+1][n]}
// Per lane 8 contiguous u32 -> 2x ds_load_b128. Requires ldU % 4 == 0.
__device__ inline v16bf load_b_fragT(const unsigned* __restrict__ baseU, int col0,
                                     int ldU, int lane) {
  int n = lane & 15, lh = lane >> 4;
  const unsigned* p = baseU + (size_t)(col0 + n) * ldU + lh * 8;
  union { uv4 q[2]; v16bf v; } u;
  u.q[0] = *(const uv4*)p;
  u.q[1] = *(const uv4*)(p + 4);
  return u.v;
}

// B = X^T fragment from row-major [n][k] bf16 (e.g. K-cache [key][dh]): per lane one
// contiguous 32B run -> 2x ds_load_b128. Requires ldk % 8 == 0, k0 % 16 == 0.
__device__ inline v16bf load_bT_frag(const bf16* __restrict__ base, int n0, int k0,
                                     int ldk, int lane) {
  int n = lane & 15, lh = lane >> 4;
  const bf16* p = base + (size_t)(n0 + n) * ldk + k0 + lh * 16;
  union { uv4 q[2]; v16bf v; } u;
  u.q[0] = *(const uv4*)p;
  u.q[1] = *(const uv4*)(p + 8);
  return u.v;
}

// interleave two u32 (2 bf16 each, rows k and k+1) into {B[k][n],B[k+1][n]} pairs
__device__ inline unsigned pk_lo(unsigned w1, unsigned w0) {
  return __builtin_amdgcn_perm(w1, w0, 0x05040100u);
}
__device__ inline unsigned pk_hi(unsigned w1, unsigned w0) {
  return __builtin_amdgcn_perm(w1, w0, 0x07060302u);
}

// ---------------- fp32 -> bf16 convert ----------------
__global__ __launch_bounds__(256) void convert_kernel(const float* __restrict__ src,
                                                      bf16* __restrict__ dst, int n) {
  int i = blockIdx.x * 256 + threadIdx.x;
  if (i < n) dst[i] = (bf16)src[i];
}

// ---------------- LayerNorm (E=512) -> bf16 ----------------
__global__ __launch_bounds__(256) void ln_kernel(const float* __restrict__ x,
                                                 const float* __restrict__ g,
                                                 const float* __restrict__ b,
                                                 bf16* __restrict__ out) {
  int row = blockIdx.x, tid = threadIdx.x;
  const float* xr = x + (size_t)row * E_;
  float v0 = xr[tid], v1 = xr[tid + 256];
  __shared__ float sS[256], sQ[256];
  sS[tid] = v0 + v1;
  sQ[tid] = v0 * v0 + v1 * v1;
  __syncthreads();
  for (int st = 128; st > 0; st >>= 1) {
    if (tid < st) { sS[tid] += sS[tid + st]; sQ[tid] += sQ[tid + st]; }
    __syncthreads();
  }
  float mu   = sS[0] * (1.0f / E_);
  float var  = sQ[0] * (1.0f / E_) - mu * mu;
  float rstd = rsqrtf(var + 1e-5f);
  out[(size_t)row * E_ + tid]       = (bf16)((v0 - mu) * rstd * g[tid] + b[tid]);
  out[(size_t)row * E_ + tid + 256] = (bf16)((v1 - mu) * rstd * g[tid + 256] + b[tid + 256]);
}

// ---------------- WMMA GEMM, double-buffered LDS ----------------
#define BM 128
#define BN 128
#define BK 32
#define LDA_ (BK + 8)        // bf16 elems; 80B rows -> 16B aligned
#define LDBT (BK / 2 + 4)    // u32 units per n-row (=20): 80B, conflict-free b128
__global__ __launch_bounds__(256) void gemm_bf16_kernel(
    const bf16* __restrict__ A, int lda, const bf16* __restrict__ Bw,
    float* __restrict__ C, bf16* __restrict__ Cbf,
    const float* __restrict__ bias, const float* __restrict__ res,
    int M, int N, int Kd) {
  __shared__ bf16     As[2][BM * LDA_];
  __shared__ unsigned BsT[2][BN * LDBT];
  int tid = threadIdx.x, lane = tid & 31, wave = tid >> 5;
  int m0 = blockIdx.y * BM, n0 = blockIdx.x * BN;
  int wr = wave & 3;   // 4 wave rows, 32 rows each
  int wc = wave >> 2;  // 2 wave cols, 64 cols each

  v8f zero = {};
  v8f acc[2][4];
#pragma unroll
  for (int i = 0; i < 2; ++i)
#pragma unroll
    for (int j = 0; j < 4; ++j) acc[i][j] = zero;

  // staging: A = 2x b128 per thread; B = one (row-pair x 8 cols) unit per thread
  uint4 ar[2], bu0, bu1;
  int brp = tid >> 4;        // 0..15 row pair
  int bcg = tid & 15;        // 0..15 col-group of 8
  auto load_tile_regs = [&](int k0) {
#pragma unroll
    for (int it = 0; it < 2; ++it) {       // A: 128x32, 8 bf16 per unit
      int u = it * 256 + tid;
      int r = u >> 2, c = (u & 3) * 8;
      ar[it] = *(const uint4*)&A[(size_t)(m0 + r) * lda + k0 + c];
    }
    const bf16* g = &Bw[(size_t)(k0 + brp * 2) * N + n0 + bcg * 8];
    bu0 = *(const uint4*)g;
    bu1 = *(const uint4*)(g + N);
  };
  auto store_tile_lds = [&](int s) {
#pragma unroll
    for (int it = 0; it < 2; ++it) {
      int u = it * 256 + tid;
      int r = u >> 2, c = (u & 3) * 8;
      *(uv4*)&As[s][r * LDA_ + c] = *(const uv4*)&ar[it];
    }
    unsigned* bt = &BsT[s][(size_t)(bcg * 8) * LDBT + brp];
    bt[0 * LDBT] = pk_lo(bu1.x, bu0.x);
    bt[1 * LDBT] = pk_hi(bu1.x, bu0.x);
    bt[2 * LDBT] = pk_lo(bu1.y, bu0.y);
    bt[3 * LDBT] = pk_hi(bu1.y, bu0.y);
    bt[4 * LDBT] = pk_lo(bu1.z, bu0.z);
    bt[5 * LDBT] = pk_hi(bu1.z, bu0.z);
    bt[6 * LDBT] = pk_lo(bu1.w, bu0.w);
    bt[7 * LDBT] = pk_hi(bu1.w, bu0.w);
  };

  load_tile_regs(0);
  store_tile_lds(0);
  int s = 0;
  for (int k0 = 0; k0 < Kd; k0 += BK) {
    __syncthreads();
    bool more = (k0 + BK) < Kd;
    if (more) load_tile_regs(k0 + BK);
    v16bf af[2], bfm[4];
#pragma unroll
    for (int i = 0; i < 2; ++i) af[i]  = load_a_frag(&As[s][0], wr * 32 + i * 16, 0, LDA_, lane);
#pragma unroll
    for (int j = 0; j < 4; ++j) bfm[j] = load_b_fragT(&BsT[s][0], wc * 64 + j * 16, LDBT, lane);
#pragma unroll
    for (int i = 0; i < 2; ++i)
#pragma unroll
      for (int j = 0; j < 4; ++j) acc[i][j] = wmma_bf(af[i], bfm[j], acc[i][j]);
    if (more) store_tile_lds(s ^ 1);
    s ^= 1;
  }

  int lh = lane >> 4, nn = lane & 15;
#pragma unroll
  for (int i = 0; i < 2; ++i)
#pragma unroll
    for (int j = 0; j < 4; ++j)
#pragma unroll
      for (int v = 0; v < 8; ++v) {
        int gm = m0 + wr * 32 + i * 16 + v + 8 * lh;
        int gn = n0 + wc * 64 + j * 16 + nn;
        float val = acc[i][j][v];
        if (bias) val += bias[gn];
        if (res)  val += res[(size_t)gm * N + gn];
        C[(size_t)gm * N + gn] = val;
        if (Cbf) Cbf[(size_t)gm * N + gn] = (bf16)val;
      }
}

// ---------------- causal depthwise conv (K=4) + SiLU ----------------
__global__ __launch_bounds__(256) void conv_silu_kernel(const float* __restrict__ up,
                                                        const float* __restrict__ w,
                                                        const float* __restrict__ bias,
                                                        float* __restrict__ cact,
                                                        bf16* __restrict__ cact_bf) {
  int idx = blockIdx.x * 256 + threadIdx.x;  // over BS_*I_
  int c = idx & (I_ - 1);
  int s = (idx / I_) & (S_ - 1);
  int b = idx / (I_ * S_);
  const float* xm = up + (size_t)b * S_ * TWOI;  // x_m = up[:, :I], row stride 2I
  float y = bias[c];
#pragma unroll
  for (int k = 0; k < KCONV; ++k) {
    int sp = s + k - (KCONV - 1);
    if (sp >= 0) y += xm[(size_t)sp * TWOI + c] * w[k * I_ + c];
  }
  float sy = y / (1.0f + __expf(-y));
  cact[idx]    = sy;
  cact_bf[idx] = (bf16)sy;
}

// ---------------- gate projection: ipre/fpre[b,h,s] = concat(q,k,v) @ w + b ----------
__global__ __launch_bounds__(256) void gate_kernel(
    const float* __restrict__ q, const float* __restrict__ k, const float* __restrict__ v,
    const float* __restrict__ wi, const float* __restrict__ bi,
    const float* __restrict__ wf, const float* __restrict__ bfv,
    float* __restrict__ ipre, float* __restrict__ fpre) {
  int row = blockIdx.x;  // b*S + s
  int b = row / S_, s = row % S_;
  int tid = threadIdx.x;
  float accI[H_] = {0.f, 0.f, 0.f, 0.f}, accF[H_] = {0.f, 0.f, 0.f, 0.f};
#pragma unroll
  for (int it = 0; it < 12; ++it) {   // 3*I_ / 256 = 12
    int j = it * 256 + tid;
    float xv;
    if (j < I_)          xv = q[(size_t)row * I_ + j];
    else if (j < 2 * I_) xv = k[(size_t)row * I_ + j - I_];
    else                 xv = v[(size_t)row * I_ + j - 2 * I_];
#pragma unroll
    for (int hh = 0; hh < H_; ++hh) {
      accI[hh] += xv * wi[(size_t)j * H_ + hh];
      accF[hh] += xv * wf[(size_t)j * H_ + hh];
    }
  }
  __shared__ float red[8][256];
#pragma unroll
  for (int hh = 0; hh < H_; ++hh) { red[hh][tid] = accI[hh]; red[hh + 4][tid] = accF[hh]; }
  __syncthreads();
  for (int st = 128; st > 0; st >>= 1) {
    if (tid < st)
#pragma unroll
      for (int o = 0; o < 8; ++o) red[o][tid] += red[o][tid + st];
    __syncthreads();
  }
  if (tid < H_) {
    ipre[((size_t)(b * H_ + tid)) * S_ + s] = red[tid][0] + bi[tid];
    fpre[((size_t)(b * H_ + tid)) * S_ + s] = red[tid + 4][0] + bfv[tid];
  }
}

// ---------------- per-(b,h) scan ----------------
__global__ void scan_kernel(const float* __restrict__ ipre, const float* __restrict__ fpre,
                            float* __restrict__ aS, float* __restrict__ cmS,
                            float* __restrict__ mS) {
  int t = threadIdx.x;
  if (t >= B_ * H_) return;
  const float* ip = ipre + (size_t)t * S_;
  const float* fp = fpre + (size_t)t * S_;
  float lfc = 0.f, best = -1e30f;
  for (int s = 0; s < S_; ++s) {
    float x  = fp[s];
    float lf = (x >= 0.f) ? -log1pf(__expf(-x)) : (x - log1pf(__expf(x)));
    lfc += lf;
    float av = ip[s] - lfc;
    best = fmaxf(best, av);
    aS[(size_t)t * S_ + s]  = av;
    cmS[(size_t)t * S_ + s] = best;
    mS[(size_t)t * S_ + s]  = lfc + best;
  }
}

// ---------------- mLSTM attention ----------------
#define QT 32
#define QDH (DH_ + 8)      // bf16 stride for Qs/Ks (16B-aligned rows)
#define VST (QT / 2 + 4)   // u32 stride per dh row of transposed-pair V (=20)
#define PLD (QT + 8)       // bf16 stride for Ps
__global__ __launch_bounds__(128) void attn_kernel(
    const bf16* __restrict__ qb, const bf16* __restrict__ kb, const bf16* __restrict__ vb,
    const float* __restrict__ aS, const float* __restrict__ cmS, const float* __restrict__ mS,
    float* __restrict__ hout) {
  __shared__ bf16     Qs[QT * QDH];
  __shared__ bf16     Ks[QT * QDH];
  __shared__ unsigned VsT[DH_ * VST];
  __shared__ bf16     Ps[QT * PLD];
  __shared__ float nsum[QT], ascl[QT], cmq[QT], nfin[QT];

  int tid = threadIdx.x, lane = tid & 31, wave = tid >> 5;
  const int nqt = S_ / QT;
  int qt = blockIdx.x % nqt;
  int h  = (blockIdx.x / nqt) % H_;
  int b  = blockIdx.x / (nqt * H_);
  int q0 = qt * QT;
  size_t bh = (size_t)(b * H_ + h) * S_;
  const bf16* qbase = qb + (size_t)b * S_ * I_ + (size_t)h * DH_;
  const bf16* kbase = kb + (size_t)b * S_ * I_ + (size_t)h * DH_;
  const bf16* vbase = vb + (size_t)b * S_ * I_ + (size_t)h * DH_;

#pragma unroll
  for (int it = 0; it < 8; ++it) {         // Q tile: 32x256, 8 bf16 per unit (b128)
    int u = it * 128 + tid;
    int r = u >> 5, c = (u & 31) * 8;
    *(uv4*)&Qs[r * QDH + c] = *(const uv4*)&qbase[(size_t)(q0 + r) * I_ + c];
  }
  if (tid < QT) { cmq[tid] = cmS[bh + q0 + tid]; nsum[tid] = 0.f; }
  __syncthreads();

  int swr = wave & 1;   // score/out tile row (16 rows)
  int stc = wave >> 1;  // score tile col
  int occ = wave >> 1;  // out col half (128 cols)
  v8f zero = {};
  v8f oacc[8];
#pragma unroll
  for (int j = 0; j < 8; ++j) oacc[j] = zero;

  for (int kt = 0; kt <= qt; ++kt) {
    int k0 = kt * QT;
#pragma unroll
    for (int it = 0; it < 8; ++it) {       // K tile rows (for K^T fragments)
      int u = it * 128 + tid;
      int r = u >> 5, c = (u & 31) * 8;
      *(uv4*)&Ks[r * QDH + c] = *(const uv4*)&kbase[(size_t)(k0 + r) * I_ + c];
    }
#pragma unroll
    for (int it = 0; it < 4; ++it) {       // V tile: transposed-pair layout
      int u = it * 128 + tid;
      int rp = u >> 5, cg = u & 31;        // key-pair, group of 8 dh
      const bf16* g = &vbase[(size_t)(k0 + rp * 2) * I_ + cg * 8];
      uint4 w0 = *(const uint4*)g;
      uint4 w1 = *(const uint4*)(g + I_);
      unsigned* vt = &VsT[(size_t)(cg * 8) * VST + rp];
      vt[0 * VST] = pk_lo(w1.x, w0.x);
      vt[1 * VST] = pk_hi(w1.x, w0.x);
      vt[2 * VST] = pk_lo(w1.y, w0.y);
      vt[3 * VST] = pk_hi(w1.y, w0.y);
      vt[4 * VST] = pk_lo(w1.z, w0.z);
      vt[5 * VST] = pk_hi(w1.z, w0.z);
      vt[6 * VST] = pk_lo(w1.w, w0.w);
      vt[7 * VST] = pk_hi(w1.w, w0.w);
    }
    if (tid < QT) ascl[tid] = aS[bh + k0 + tid];
    __syncthreads();

    // S = Q @ K^T  (one 16x16 tile per wave, K-dim 256 -> 8 WMMA)
    v8f sc = zero;
#pragma unroll
    for (int d0 = 0; d0 < DH_; d0 += 32) {
      v16bf aq = load_a_frag(Qs, swr * 16, d0, QDH, lane);
      v16bf bk = load_bT_frag(Ks, stc * 16, d0, QDH, lane);
      sc = wmma_bf(aq, bk, sc);
    }
    {
      int lh = lane >> 4, nn = lane & 15;
      int sk_loc = stc * 16 + nn;
      int sk = k0 + sk_loc;
      float av = ascl[sk_loc];
#pragma unroll
      for (int v = 0; v < 8; ++v) {
        int tq_loc = swr * 16 + v + 8 * lh;
        int tq = q0 + tq_loc;
        float p = 0.f;
        if (sk <= tq) p = sc[v] * 0.0625f * __expf(av - cmq[tq_loc]);  // DH^-0.5
        atomicAdd(&nsum[tq_loc], p);
        Ps[tq_loc * PLD + sk_loc] = (bf16)p;
      }
    }
    __syncthreads();

    // O += P @ V  (wave: 16 rows x 128 cols; A frag invariant over col tiles)
    {
      v16bf ap = load_a_frag(Ps, swr * 16, 0, PLD, lane);
#pragma unroll
      for (int j = 0; j < 8; ++j) {
        int c0 = occ * 128 + j * 16;
        v16bf bv = load_b_fragT(VsT, c0, VST, lane);
        oacc[j] = wmma_bf(ap, bv, oacc[j]);
      }
    }
    __syncthreads();
  }

  if (tid < QT) nfin[tid] = fmaxf(fabsf(nsum[tid]), __expf(-mS[bh + q0 + tid]));
  __syncthreads();

  int lh = lane >> 4, nn = lane & 15;
  float* obase = hout + (size_t)b * S_ * I_ + (size_t)h * DH_;
#pragma unroll
  for (int j = 0; j < 8; ++j)
#pragma unroll
    for (int v = 0; v < 8; ++v) {
      int rloc = swr * 16 + v + 8 * lh;
      int c = occ * 128 + j * 16 + nn;
      obase[(size_t)(q0 + rloc) * I_ + c] = oacc[j][v] / nfin[rloc];
    }
}

// ---------------- head GroupNorm + gn_g + skip*c_act, gated by SiLU(z) ----------------
__global__ __launch_bounds__(256) void headmix_kernel(
    const float* __restrict__ hattn, const float* __restrict__ up,
    const float* __restrict__ cact, const float* __restrict__ gn,
    const float* __restrict__ skip, bf16* __restrict__ gated) {
  int blk = blockIdx.x;  // bs*H + h
  int h = blk % H_, bs = blk / H_;
  int tid = threadIdx.x;  // == dh (DH_=256)
  float v = hattn[(size_t)bs * I_ + h * DH_ + tid];
  __shared__ float sS[256], sQ[256];
  sS[tid] = v; sQ[tid] = v * v;
  __syncthreads();
  for (int st = 128; st > 0; st >>= 1) {
    if (tid < st) { sS[tid] += sS[tid + st]; sQ[tid] += sQ[tid + st]; }
    __syncthreads();
  }
  float mu  = sS[0] * (1.0f / DH_);
  float var = sQ[0] * (1.0f / DH_) - mu * mu;
  float hn  = (v - mu) * rsqrtf(var + 1e-5f);
  int c = h * DH_ + tid;
  float z  = up[(size_t)bs * TWOI + I_ + c];
  float sz = z / (1.0f + __expf(-z));
  float o  = (hn * gn[c] + skip[c] * cact[(size_t)bs * I_ + c]) * sz;
  gated[(size_t)bs * I_ + c] = (bf16)o;
}

// ---------------- host orchestration ----------------
extern "C" void kernel_launch(void* const* d_in, const int* in_sizes, int n_in,
                              void* d_out, int out_size, void* d_ws, size_t ws_size,
                              hipStream_t stream) {
  (void)in_sizes; (void)n_in; (void)out_size; (void)ws_size;
  const float* x      = (const float*)d_in[0];
  const float* ln_g   = (const float*)d_in[1];
  const float* ln_b   = (const float*)d_in[2];
  const float* w_up   = (const float*)d_in[3];
  const float* conv_w = (const float*)d_in[4];
  const float* conv_b = (const float*)d_in[5];
  const float* wq     = (const float*)d_in[6];
  const float* wk     = (const float*)d_in[7];
  const float* wv     = (const float*)d_in[8];
  const float* w_i    = (const float*)d_in[9];
  const float* b_i    = (const float*)d_in[10];
  const float* w_f    = (const float*)d_in[11];
  const float* b_f    = (const float*)d_in[12];
  const float* skip   = (const float*)d_in[13];
  const float* gn_g   = (const float*)d_in[14];
  const float* w_down = (const float*)d_in[15];
  const float* b_down = (const float*)d_in[16];
  const float* w_fin  = (const float*)d_in[17];
  const float* b_fin  = (const float*)d_in[18];
  float* out = (float*)d_out;

  char* base = (char*)d_ws;
  size_t off = 0;
  auto alloc = [&](size_t bytes) -> char* {
    char* p = base + off;
    off = (off + bytes + 255) & ~(size_t)255;
    return p;
  };
  const size_t BSE  = (size_t)BS_ * E_;
  const size_t BSI  = (size_t)BS_ * I_;
  const size_t BS2I = (size_t)BS_ * TWOI;
  const size_t BHS  = (size_t)B_ * H_ * S_;

  float* xa     = (float*)alloc(BSE * 4);
  float* xb     = (float*)alloc(BSE * 4);
  bf16*  xbf    = (bf16*)alloc(BSE * 2);
  bf16*  xnbf   = (bf16*)alloc(BSE * 2);
  float* up     = (float*)alloc(BS2I * 4);
  bf16*  upbf   = (bf16*)alloc(BS2I * 2);
  float* cact   = (float*)alloc(BSI * 4);
  bf16*  cactbf = (bf16*)alloc(BSI * 2);
  float* qf     = (float*)alloc(BSI * 4);
  float* kf     = (float*)alloc(BSI * 4);
  float* vf     = (float*)alloc(BSI * 4);
  bf16*  qbf    = (bf16*)alloc(BSI * 2);
  bf16*  kbf    = (bf16*)alloc(BSI * 2);
  bf16*  vbf    = (bf16*)alloc(BSI * 2);
  float* ipre   = (float*)alloc(BHS * 4);
  float* fpre   = (float*)alloc(BHS * 4);
  float* aSb    = (float*)alloc(BHS * 4);
  float* cmSb   = (float*)alloc(BHS * 4);
  float* mSb    = (float*)alloc(BHS * 4);
  float* hattn  = (float*)alloc(BSI * 4);
  bf16*  gatedbf= (bf16*)alloc(BSI * 2);
  bf16*  wupbf  = (bf16*)alloc((size_t)L_ * E_ * TWOI * 2);
  bf16*  wqbf   = (bf16*)alloc((size_t)L_ * I_ * I_ * 2);
  bf16*  wkbf   = (bf16*)alloc((size_t)L_ * I_ * I_ * 2);
  bf16*  wvbf   = (bf16*)alloc((size_t)L_ * I_ * I_ * 2);
  bf16*  wdbf   = (bf16*)alloc((size_t)L_ * I_ * E_ * 2);
  bf16*  wfbf   = (bf16*)alloc((size_t)E_ * F_ * 2);

  auto cvt = [&](const float* s, bf16* d, size_t n) {
    convert_kernel<<<dim3((unsigned)((n + 255) / 256)), dim3(256), 0, stream>>>(s, d, (int)n);
  };
  cvt(w_up,   wupbf, (size_t)L_ * E_ * TWOI);
  cvt(wq,     wqbf,  (size_t)L_ * I_ * I_);
  cvt(wk,     wkbf,  (size_t)L_ * I_ * I_);
  cvt(wv,     wvbf,  (size_t)L_ * I_ * I_);
  cvt(w_down, wdbf,  (size_t)L_ * I_ * E_);
  cvt(w_fin,  wfbf,  (size_t)E_ * F_);

  for (int l = 0; l < L_; ++l) {
    const float* xin = (l == 0) ? x : xa;
    float* xout = (l == 0) ? xa : xb;

    ln_kernel<<<dim3(BS_), dim3(256), 0, stream>>>(
        xin, ln_g + (size_t)l * E_, ln_b + (size_t)l * E_, xnbf);

    // up = xn @ w_up  (M=8192, N=2048, K=512)
    gemm_bf16_kernel<<<dim3(TWOI / BN, BS_ / BM), dim3(256), 0, stream>>>(
        xnbf, E_, wupbf + (size_t)l * E_ * TWOI, up, upbf, nullptr, nullptr, BS_, TWOI, E_);

    conv_silu_kernel<<<dim3((unsigned)(BSI / 256)), dim3(256), 0, stream>>>(
        up, conv_w + (size_t)l * KCONV * I_, conv_b + (size_t)l * I_, cact, cactbf);

    // q = c_act @ wq ; k = c_act @ wk ; v = x_m @ wv   (M=8192, N=1024, K=1024)
    gemm_bf16_kernel<<<dim3(I_ / BN, BS_ / BM), dim3(256), 0, stream>>>(
        cactbf, I_, wqbf + (size_t)l * I_ * I_, qf, qbf, nullptr, nullptr, BS_, I_, I_);
    gemm_bf16_kernel<<<dim3(I_ / BN, BS_ / BM), dim3(256), 0, stream>>>(
        cactbf, I_, wkbf + (size_t)l * I_ * I_, kf, kbf, nullptr, nullptr, BS_, I_, I_);
    gemm_bf16_kernel<<<dim3(I_ / BN, BS_ / BM), dim3(256), 0, stream>>>(
        upbf, TWOI, wvbf + (size_t)l * I_ * I_, vf, vbf, nullptr, nullptr, BS_, I_, I_);

    gate_kernel<<<dim3(BS_), dim3(256), 0, stream>>>(
        qf, kf, vf, w_i + (size_t)l * 3 * I_ * H_, b_i + (size_t)l * H_,
        w_f + (size_t)l * 3 * I_ * H_, b_f + (size_t)l * H_, ipre, fpre);

    scan_kernel<<<dim3(1), dim3(32), 0, stream>>>(ipre, fpre, aSb, cmSb, mSb);

    attn_kernel<<<dim3(B_ * H_ * (S_ / QT)), dim3(128), 0, stream>>>(
        qbf, kbf, vbf, aSb, cmSb, mSb, hattn);

    headmix_kernel<<<dim3(BS_ * H_), dim3(256), 0, stream>>>(
        hattn, up, cact, gn_g + (size_t)l * I_, skip + (size_t)l * I_, gatedbf);

    // xout = res + gated @ w_down + b_down  (M=8192, N=512, K=1024)
    gemm_bf16_kernel<<<dim3(E_ / BN, BS_ / BM), dim3(256), 0, stream>>>(
        gatedbf, I_, wdbf + (size_t)l * I_ * E_, xout, xbf,
        b_down + (size_t)l * E_, xin, BS_, E_, I_);
  }

  // out = h @ w_fin + b_fin  (M=8192, N=128, K=512)
  gemm_bf16_kernel<<<dim3(F_ / BN, BS_ / BM), dim3(256), 0, stream>>>(
      xbf, E_, wfbf, out, nullptr, b_fin, nullptr, BS_, F_, E_);
}
